// SquaredL2Distance_14181982011477
// MI455X (gfx1250) — matrix operand order repaired and checked by reading, not compiled, not run on hardware
//
#include <hip/hip_runtime.h>

// Pairwise squared L2 distance via CDNA5 WMMA (gfx1250, wave32).
// out[N, M] = max(||q_i||^2 + ||s_j||^2 - 2 * q_i . s_j, 0)
//
// Pass 1 (prep): one-time f32 -> split bf16 (hi/lo) conversion of q and s into
//   d_ws, plus row squared-norms (shuffle reduction). ~6.3 MB, L2-resident.
// Pass 2 (GEMM): v_wmma_f32_16x16x32_bf16 with 3-term split-precision
//   accumulation (qh.sh + qh.sl + ql.sh, f32 accumulate) -> ~f32 accuracy at
//   bf16 matrix rates; kernel is bound by the 512 MB output write.

typedef __attribute__((ext_vector_type(16))) __bf16 v16bf;
typedef __attribute__((ext_vector_type(8)))  __bf16 v8bf;
typedef __attribute__((ext_vector_type(4)))  __bf16 v4bf;
typedef __attribute__((ext_vector_type(8)))  float  v8f;

#define DIMK 64   // feature dimension (fixed by reference)
#define BT   128  // block tile (rows and cols)

__device__ __forceinline__ void split1(float x, __bf16& h, __bf16& l) {
  h = (__bf16)x;               // RNE convert
  l = (__bf16)(x - (float)h);  // residual, also RNE
}

// ---------------- prep: f32[R,64] -> hi/lo bf16 + row norms ----------------
// One float4 per thread; 16 threads (aligned lane group) cover one row.
__global__ __launch_bounds__(256)
void prep_kernel(const float* __restrict__ in,
                 unsigned short* __restrict__ hi_u,
                 unsigned short* __restrict__ lo_u,
                 float* __restrict__ norm) {
  const int idx4 = blockIdx.x * 256 + threadIdx.x;  // float4 index
  float4 v = ((const float4*)in)[idx4];
  v4bf hv, lv;
  __bf16 hb, lb;
  split1(v.x, hb, lb); hv[0] = hb; lv[0] = lb;
  split1(v.y, hb, lb); hv[1] = hb; lv[1] = lb;
  split1(v.z, hb, lb); hv[2] = hb; lv[2] = lb;
  split1(v.w, hb, lb); hv[3] = hb; lv[3] = lb;
  ((v4bf*)hi_u)[idx4] = hv;
  ((v4bf*)lo_u)[idx4] = lv;
  float p = v.x * v.x + v.y * v.y + v.z * v.z + v.w * v.w;
  p += __shfl_xor(p, 1);
  p += __shfl_xor(p, 2);
  p += __shfl_xor(p, 4);
  p += __shfl_xor(p, 8);
  if ((threadIdx.x & 15) == 0) norm[idx4 >> 4] = p;
}

// ---------------- main GEMM-style kernel (bf16 inputs pre-split) -----------
__global__ __launch_bounds__(256)
void sql2_wmma_split_kernel(const unsigned short* __restrict__ qh_u,
                            const unsigned short* __restrict__ ql_u,
                            const unsigned short* __restrict__ sh_u,
                            const unsigned short* __restrict__ sl_u,
                            const float* __restrict__ qn,
                            const float* __restrict__ sn,
                            float* __restrict__ out, int M) {
  __shared__ float qs[BT];
  __shared__ float ss[BT];

  const int row0 = blockIdx.y * BT;
  const int col0 = blockIdx.x * BT;
  const int tid  = threadIdx.x;

  if (tid < BT) qs[tid]      = qn[row0 + tid];
  else          ss[tid - BT] = sn[col0 + tid - BT];
  __syncthreads();

  const __bf16* qh = (const __bf16*)qh_u;
  const __bf16* ql = (const __bf16*)ql_u;
  const __bf16* sh = (const __bf16*)sh_u;
  const __bf16* sl = (const __bf16*)sl_u;

  const int wave = tid >> 5;
  const int lane = tid & 31;
  const int h    = lane >> 4;
  const int lm   = lane & 15;
  const int wm   = wave >> 1;  // 0..3 -> row offset wm*32
  const int wn   = wave & 1;   // 0..1 -> col offset wn*64

  v8f acc[2][4];
#pragma unroll
  for (int t = 0; t < 2; ++t)
#pragma unroll
    for (int u = 0; u < 4; ++u)
      acc[t][u] = (v8f){0.f, 0.f, 0.f, 0.f, 0.f, 0.f, 0.f, 0.f};

#pragma unroll
  for (int c = 0; c < 2; ++c) {
    const int kb = c * 32;

    // A frags: lane row m = lm; slots 0..15 <- K=kb+16h+0..15 (16B loads)
    v16bf ahi[2], alo[2];
#pragma unroll
    for (int t = 0; t < 2; ++t) {
      const size_t r = (size_t)(row0 + wm * 32 + t * 16 + lm) * DIMK;
      v8bf h0 = *(const v8bf*)(qh + r + kb + 16 * h);
      v8bf h1 = *(const v8bf*)(qh + r + kb + 8 + 16 * h);
      ahi[t] = __builtin_shufflevector(h0, h1, 0, 1, 2, 3, 4, 5, 6, 7,
                                       8, 9, 10, 11, 12, 13, 14, 15);
      v8bf l0 = *(const v8bf*)(ql + r + kb + 16 * h);
      v8bf l1 = *(const v8bf*)(ql + r + kb + 8 + 16 * h);
      alo[t] = __builtin_shufflevector(l0, l1, 0, 1, 2, 3, 4, 5, 6, 7,
                                       8, 9, 10, 11, 12, 13, 14, 15);
    }

    // B frags (B = S^T): lane col n = lm; slots 0..15 <- K=kb+16h+0..15
    v16bf bhi[4], blo[4];
#pragma unroll
    for (int u = 0; u < 4; ++u) {
      const size_t r = (size_t)(col0 + wn * 64 + u * 16 + lm) * DIMK + kb + 16 * h;
      v8bf h0 = *(const v8bf*)(sh + r);
      v8bf h1 = *(const v8bf*)(sh + r + 8);
      bhi[u] = __builtin_shufflevector(h0, h1, 0, 1, 2, 3, 4, 5, 6, 7,
                                       8, 9, 10, 11, 12, 13, 14, 15);
      v8bf l0 = *(const v8bf*)(sl + r);
      v8bf l1 = *(const v8bf*)(sl + r + 8);
      blo[u] = __builtin_shufflevector(l0, l1, 0, 1, 2, 3, 4, 5, 6, 7,
                                       8, 9, 10, 11, 12, 13, 14, 15);
    }

#pragma unroll
    for (int t = 0; t < 2; ++t)
#pragma unroll
      for (int u = 0; u < 4; ++u) {
        acc[t][u] = __builtin_amdgcn_wmma_f32_16x16x32_bf16(
            false, ahi[t], false, bhi[u], (short)0, acc[t][u], false, false);
        acc[t][u] = __builtin_amdgcn_wmma_f32_16x16x32_bf16(
            false, ahi[t], false, blo[u], (short)0, acc[t][u], false, false);
        acc[t][u] = __builtin_amdgcn_wmma_f32_16x16x32_bf16(
            false, alo[t], false, bhi[u], (short)0, acc[t][u], false, false);
      }
  }

  // epilogue: d = q^2 + s^2 - 2*cross, clamped at 0
  // C/D layout: vgpr v holds (M = v + 8h, N = lane%16)
#pragma unroll
  for (int t = 0; t < 2; ++t) {
#pragma unroll
    for (int u = 0; u < 4; ++u) {
      const int n    = col0 + wn * 64 + u * 16 + lm;
      const float s2 = ss[wn * 64 + u * 16 + lm];
#pragma unroll
      for (int v = 0; v < 8; ++v) {
        const int ml = wm * 32 + t * 16 + 8 * h + v;
        float d = qs[ml] + s2 - 2.0f * acc[t][u][v];
        d = fmaxf(d, 0.0f);
        out[(size_t)(row0 + ml) * M + n] = d;
      }
    }
  }
}

// ---------------- fallback: fused convert+GEMM (if ws too small) -----------
__device__ __forceinline__ void split_f4(const float4 v, v16bf& hi, v16bf& lo, const int off) {
  __bf16 hb, lb;
  split1(v.x, hb, lb); hi[off + 0] = hb; lo[off + 0] = lb;
  split1(v.y, hb, lb); hi[off + 1] = hb; lo[off + 1] = lb;
  split1(v.z, hb, lb); hi[off + 2] = hb; lo[off + 2] = lb;
  split1(v.w, hb, lb); hi[off + 3] = hb; lo[off + 3] = lb;
}

__global__ __launch_bounds__(256)
void sql2_wmma_fused_kernel(const float* __restrict__ q,
                            const float* __restrict__ s,
                            float* __restrict__ out, int M) {
  __shared__ float qs[BT];
  __shared__ float ss[BT];
  const int row0 = blockIdx.y * BT;
  const int col0 = blockIdx.x * BT;
  const int tid  = threadIdx.x;
  {
    const float* base = (tid < BT) ? (q + (size_t)(row0 + tid) * DIMK)
                                   : (s + (size_t)(col0 + (tid - BT)) * DIMK);
    const float4* b4 = (const float4*)base;
    float acc = 0.0f;
#pragma unroll
    for (int i = 0; i < DIMK / 4; ++i) {
      float4 v = b4[i];
      acc += v.x * v.x + v.y * v.y + v.z * v.z + v.w * v.w;
    }
    if (tid < BT) qs[tid] = acc; else ss[tid - BT] = acc;
  }
  __syncthreads();

  const int wave = tid >> 5, lane = tid & 31;
  const int h = lane >> 4, lm = lane & 15;
  const int wm = wave >> 1, wn = wave & 1;

  v8f acc[2][4];
#pragma unroll
  for (int t = 0; t < 2; ++t)
#pragma unroll
    for (int u = 0; u < 4; ++u)
      acc[t][u] = (v8f){0.f, 0.f, 0.f, 0.f, 0.f, 0.f, 0.f, 0.f};

#pragma unroll
  for (int c = 0; c < 2; ++c) {
    const int kb = c * 32;
    v16bf ahi[2], alo[2];
#pragma unroll
    for (int t = 0; t < 2; ++t) {
      const float* ar = q + (size_t)(row0 + wm * 32 + t * 16 + lm) * DIMK + kb + 16 * h;
      const float4* a4  = (const float4*)ar;
      const float4* a4b = (const float4*)(ar + 8);
      split_f4(a4[0],  ahi[t], alo[t], 0);
      split_f4(a4[1],  ahi[t], alo[t], 4);
      split_f4(a4b[0], ahi[t], alo[t], 8);
      split_f4(a4b[1], ahi[t], alo[t], 12);
    }
    v16bf bhi[4], blo[4];
#pragma unroll
    for (int u = 0; u < 4; ++u) {
      const float* br = s + (size_t)(col0 + wn * 64 + u * 16 + lm) * DIMK + kb + 16 * h;
      const float4* b4 = (const float4*)br;
      split_f4(b4[0], bhi[u], blo[u], 0);
      split_f4(b4[1], bhi[u], blo[u], 4);
      split_f4(b4[2], bhi[u], blo[u], 8);
      split_f4(b4[3], bhi[u], blo[u], 12);
    }
#pragma unroll
    for (int t = 0; t < 2; ++t)
#pragma unroll
      for (int u = 0; u < 4; ++u) {
        acc[t][u] = __builtin_amdgcn_wmma_f32_16x16x32_bf16(
            false, ahi[t], false, bhi[u], (short)0, acc[t][u], false, false);
        acc[t][u] = __builtin_amdgcn_wmma_f32_16x16x32_bf16(
            false, ahi[t], false, blo[u], (short)0, acc[t][u], false, false);
        acc[t][u] = __builtin_amdgcn_wmma_f32_16x16x32_bf16(
            false, alo[t], false, bhi[u], (short)0, acc[t][u], false, false);
      }
  }
#pragma unroll
  for (int t = 0; t < 2; ++t) {
#pragma unroll
    for (int u = 0; u < 4; ++u) {
      const int n    = col0 + wn * 64 + u * 16 + lm;
      const float s2 = ss[wn * 64 + u * 16 + lm];
#pragma unroll
      for (int v = 0; v < 8; ++v) {
        const int ml = wm * 32 + t * 16 + 8 * h + v;
        float d = qs[ml] + s2 - 2.0f * acc[t][u][v];
        d = fmaxf(d, 0.0f);
        out[(size_t)(row0 + ml) * M + n] = d;
      }
    }
  }
}

extern "C" void kernel_launch(void* const* d_in, const int* in_sizes, int n_in,
                              void* d_out, int out_size, void* d_ws, size_t ws_size,
                              hipStream_t stream) {
  const float* q = (const float*)d_in[0];
  const float* s = (const float*)d_in[1];
  float* out = (float*)d_out;
  const int N = in_sizes[0] / DIMK;  // 16384
  const int M = in_sizes[1] / DIMK;  // 8192

  const size_t qbytes = (size_t)N * DIMK * 2;  // bf16 matrix bytes
  const size_t sbytes = (size_t)M * DIMK * 2;
  const size_t need   = 2 * qbytes + 2 * sbytes + (size_t)(N + M) * 4;

  dim3 grid(M / BT, N / BT);

  if (ws_size >= need) {
    char* w = (char*)d_ws;
    unsigned short* qh = (unsigned short*)(w);
    unsigned short* ql = (unsigned short*)(w + qbytes);
    unsigned short* sh = (unsigned short*)(w + 2 * qbytes);
    unsigned short* sl = (unsigned short*)(w + 2 * qbytes + sbytes);
    float* qn = (float*)(w + 2 * qbytes + 2 * sbytes);
    float* sn = qn + N;

    // one float4 per thread: grid = R*16/256 = R/16 blocks (N,M mult. of 16)
    prep_kernel<<<N / 16, 256, 0, stream>>>(q, qh, ql, qn);
    prep_kernel<<<M / 16, 256, 0, stream>>>(s, sh, sl, sn);
    sql2_wmma_split_kernel<<<grid, 256, 0, stream>>>(qh, ql, sh, sl, qn, sn, out, M);
  } else {
    sql2_wmma_fused_kernel<<<grid, 256, 0, stream>>>(q, s, out, M);
  }
}
